// BondCenteredTensorMomentDescriptor_50251117363955
// MI455X (gfx1250) — matrix operand order repaired and checked by reading, not compiled, not run on hardware
//
#include <hip/hip_runtime.h>

typedef __attribute__((ext_vector_type(16))) _Float16 v16h;
typedef __attribute__((ext_vector_type(8)))  float    v8f;
typedef __attribute__((ext_vector_type(4)))  float    v4f;

#if __has_builtin(__builtin_amdgcn_global_store_async_from_lds_b128)
#define HAVE_ASYNC_LDS 1
typedef int v4i __attribute__((vector_size(16)));
typedef __attribute__((address_space(1))) v4i g_v4i;   // global int4
typedef __attribute__((address_space(3))) v4i l_v4i;   // LDS int4
#endif

// ====================== compile-time Clebsch-Gordan ======================
namespace cgc {

constexpr double FACT[13] = {1.,1.,2.,6.,24.,120.,720.,5040.,40320.,
                             362880.,3628800.,39916800.,479001600.};

constexpr double csqrt(double x){
  if (x <= 0.0) return 0.0;
  double g = x > 1.0 ? x : 1.0;
  for (int i = 0; i < 64; i++) g = 0.5*(g + x/g);
  return g;
}

constexpr double cg(int j1,int m1,int j2,int m2,int j3,int m3){
  if (m1 + m2 != m3) return 0.0;
  int lo = j1 - j2; if (lo < 0) lo = -lo;
  if (j3 < lo || j3 > j1 + j2) return 0.0;
  double pre = csqrt((2*j3+1)*FACT[j1+j2-j3]*FACT[j1-j2+j3]*FACT[-j1+j2+j3]/FACT[j1+j2+j3+1]);
  pre *= csqrt(FACT[j1+m1]*FACT[j1-m1]*FACT[j2+m2]*FACT[j2-m2]*FACT[j3+m3]*FACT[j3-m3]);
  int kmin = 0;
  if (j2-j3-m1 > kmin) kmin = j2-j3-m1;
  if (j1-j3+m2 > kmin) kmin = j1-j3+m2;
  int kmax = j1+j2-j3;
  if (j1-m1 < kmax) kmax = j1-m1;
  if (j2+m2 < kmax) kmax = j2+m2;
  double s = 0.0;
  for (int k = kmin; k <= kmax; k++){
    double t = 1.0/(FACT[k]*FACT[j1+j2-j3-k]*FACT[j1-m1-k]*FACT[j2+m2-k]*FACT[j3-j2+m1+k]*FACT[j3-j1-m2+k]);
    s += (k & 1) ? -t : t;
  }
  return pre*s;
}

struct CD { double re, im; };
constexpr CD cmul(CD a, CD b){ return {a.re*b.re - a.im*b.im, a.re*b.im + a.im*b.re}; }

constexpr CD uent(int l, int r, int c){
  constexpr double s2 = 0.7071067811865475244;
  int mr = r - l;
  if (mr == 0) return (c == l) ? CD{1.0, 0.0} : CD{0.0, 0.0};
  if (mr > 0){
    int m = mr; double sg = (m & 1) ? -1.0 : 1.0;
    if (c == l + m) return {sg*s2, 0.0};
    if (c == l - m) return {s2, 0.0};
    return {0.0, 0.0};
  }
  int m = -mr; double sg = (m & 1) ? -1.0 : 1.0;
  if (c == l + m) return {0.0, -sg*s2};
  if (c == l - m) return {0.0, s2};
  return {0.0, 0.0};
}
constexpr CD uentc(int l, int r, int c){ CD u = uent(l, r, c); return {u.re, -u.im}; }

template<int L1,int L2,int L3>
struct CGTab {
  static constexpr int N1 = 2*L1+1, N2 = 2*L2+1;
  struct T { double v[N1*N2]; };
  static constexpr T make(){
    T t{};
    for (int a = 0; a < N1; a++)
      for (int b = 0; b < N2; b++){
        int m3 = (a - L1) + (b - L2);
        t.v[a*N2 + b] = (m3 >= -L3 && m3 <= L3) ? cg(L1, a-L1, L2, b-L2, L3, m3) : 0.0;
      }
    return t;
  }
  static constexpr T tab = make();
};

template<int L1,int L2,int L3>
struct RCG {
  static constexpr int N1 = 2*L1+1, N2 = 2*L2+1, N3 = 2*L3+1;
  struct T { float v[N1*N2*N3]; };
  static constexpr T make(){
    T t{};
    for (int a = 0; a < N1; a++)
      for (int b = 0; b < N2; b++)
        for (int c = 0; c < N3; c++){
          double sre = 0.0, sim = 0.0;
          int m1s[2] = {a, 2*L1 - a}; int n1 = (m1s[0] == m1s[1]) ? 1 : 2;
          int m2s[2] = {b, 2*L2 - b}; int n2 = (m2s[0] == m2s[1]) ? 1 : 2;
          for (int i = 0; i < n1; i++)
            for (int j = 0; j < n2; j++){
              int m1 = m1s[i], m2 = m2s[j];
              int k = (m1 - L1) + (m2 - L2) + L3;
              if (k < 0 || k >= N3) continue;
              if (k != c && k != 2*L3 - c) continue;
              CD term = cmul(cmul(uentc(L1, a, m1), uentc(L2, b, m2)), uent(L3, c, k));
              double g = CGTab<L1,L2,L3>::tab.v[m1*N2 + m2];
              sre += term.re*g; sim += term.im*g;
            }
          double o = sre + sim;
          if (o < 1e-12 && o > -1e-12) o = 0.0;
          t.v[(a*N2 + b)*N3 + c] = (float)o;
        }
    return t;
  }
  static constexpr T tab = make();
};

} // namespace cgc

// ====================== static_for ======================
template<int I> struct IC { static constexpr int v = I; };
template<int I,int N,class F>
__device__ __forceinline__ void static_for(F&& f){
  if constexpr (I < N){ f(IC<I>{}); static_for<I+1,N>(f); }
}

// ====================== tensor product path (fully unrolled, zeros elided) ==
template<int PI,int L1,int L2,int L3>
__device__ __forceinline__ void tp_path(const float* __restrict__ Wtp, int f,
                                        const float (&a)[9],
                                        const float (&b)[2][25],
                                        float (&o)[2][25]){
  constexpr int odd = L1 & 1;
  float w0 = Wtp[PI*32 + f];        // Wtp[path][0][f]
  float w1 = Wtp[PI*32 + 16 + f];   // Wtp[path][1][f]
  static_for<0, 2*L1+1>([&](auto NI){
    constexpr int n = decltype(NI)::v;
    float aw0 = a[L1*L1 + n] * w0;
    float aw1 = a[L1*L1 + n] * w1;
    static_for<0, 2*L2+1>([&](auto MI){
      constexpr int m = decltype(MI)::v;
      static_for<0, 2*L3+1>([&](auto KI){
        constexpr int k = decltype(KI)::v;
        constexpr float cv = cgc::RCG<L1,L2,L3>::tab.v[(n*(2*L2+1) + m)*(2*L3+1) + k];
        if constexpr (cv != 0.0f){
          o[odd    ][L3*L3 + k] += (cv*aw0) * b[0][L2*L2 + m];
          o[1 - odd][L3*L3 + k] += (cv*aw1) * b[1][L2*L2 + m];
        }
      });
    });
  });
}

// ====================== edense via WMMA ======================
// Within one wave: lane L = 16h + c : A row = bond c, K-half = h. K padded 16->32.
// Wave handles one parity p and l in {lpar, lpar+2, ...}.
template<bool RES>
__device__ __forceinline__ void edense_y(float* __restrict__ sy,
                                         const float* __restrict__ W,
                                         const float* __restrict__ bvec,
                                         int c, int h, int p, int lpar){
  for (int l = lpar; l < 5; l += 2){
    v16h B;
    #pragma unroll
    for (int i = 0; i < 16; i++){
      float wv = (i < 8) ? W[((p*5 + l)*16 + 8*h + i)*16 + c] : 0.f;   // W[p][l][f=8h+i][g=c]
      B[i] = (_Float16)wv;
    }
    const int mlo = l*l, mhi = (l+1)*(l+1);
    for (int m = mlo; m < mhi; m++){
      const float* ap = &sy[((c*2 + p)*25 + m)*16 + 8*h];
      v16h A;
      #pragma unroll
      for (int i = 0; i < 8; i++){ A[i] = (_Float16)ap[i]; A[8+i] = (_Float16)0.f; }
      v8f C;
      #pragma unroll
      for (int r = 0; r < 8; r++){
        float cval = RES ? sy[(((r + 8*h)*2 + p)*25 + m)*16 + c] : 0.f;
        C[r] = cval;
      }
      if (p == 0 && m == 0){
        float bv = bvec[c];
        #pragma unroll
        for (int r = 0; r < 8; r++) C[r] += bv;
      }
      v8f D = __builtin_amdgcn_wmma_f32_16x16x32_f16(false, A, false, B,
                                                     (short)0, C, false, false);
      #pragma unroll
      for (int r = 0; r < 8; r++)
        sy[(((r + 8*h)*2 + p)*25 + m)*16 + c] = D[r];
    }
  }
}

// ====================== fused kernel: 4 waves = 16 bonds ======================
__global__ __launch_bounds__(128)
__attribute__((amdgpu_waves_per_eu(4)))
void bctmd_kernel(const float* __restrict__ desc, const int* __restrict__ nidx,
                  const float* __restrict__ disp,
                  const float* __restrict__ W1, const float* __restrict__ b1,
                  const float* __restrict__ gamma, const float* __restrict__ beta,
                  const float* __restrict__ W2, const float* __restrict__ b2,
                  const float* __restrict__ Wb, const float* __restrict__ bb,
                  const float* __restrict__ Wtp, float* __restrict__ outp,
                  int nBonds)
{
  __shared__ alignas(16) float sy[16*800];     // 16 bonds x 2p x 25m x 16f (51200 B)
  __shared__ alignas(16) float sbed[16*144];   // 16 bonds x 9n x 16g       ( 9216 B)

  const int t = threadIdx.x;
  const int c = t & 15;            // within-wave WMMA column / bond row
  const int h = (t >> 4) & 1;      // within-wave K-half
  const int w = t >> 5;            // wave id 0..3
  const int e0 = blockIdx.x * 16;

  // ---- gather: y0 = desc[i] + desc[j]  (L2-resident table) ----
  {
    v4f* sy4 = (v4f*)sy;
    const v4f* d4 = (const v4f*)desc;
    for (int tt = t; tt < 3200; tt += 128){     // 16 bonds * 200 float4
      int e = tt / 200, q = tt - e*200;
      int bond = e0 + e;
      v4f v = {0.f, 0.f, 0.f, 0.f};
      if (bond < nBonds){
        int ia = nidx[2*bond], ja = nidx[2*bond + 1];
        v = d4[ia*200 + q] + d4[ja*200 + q];
      }
      sy4[tt] = v;
    }
  }
  __syncthreads();

  // ---- edense #1 (WMMA): wave w -> parity w>>1, l in {w&1, w&1+2, ...} ----
  edense_y<false>(sy, W1, b1, c, h, w >> 1, w & 1);
  __syncthreads();

  // ---- eln + emish ; thread handles (bond e, parity p, f-quarter q) ----
  {
    const int e = t >> 3;          // 0..15
    const int p = (t >> 2) & 1;
    const int q = t & 3;           // f = 4q .. 4q+3
    float* base = &sy[(e*2 + p)*400];

    float acc[5] = {0.f,0.f,0.f,0.f,0.f};
    #pragma unroll
    for (int l = 0; l < 5; l++){
      for (int m = l*l; m < (l+1)*(l+1); m++){
        v4f v = ((const v4f*)(base + m*16))[q];
        acc[l] += v.x*v.x + v.y*v.y + v.z*v.z + v.w*v.w;
      }
    }
    float invr[5];
    #pragma unroll
    for (int l = 0; l < 5; l++){
      float a2 = acc[l];
      a2 += __shfl_xor(a2, 1);
      a2 += __shfl_xor(a2, 2);
      invr[l] = 1.f/sqrtf(a2*(1.f/16.f) + 1e-5f);
    }

    // scalar channel (p=0,l=0,m=0): LayerNorm then mish (+derivative)
    v4f sv = ((const v4f*)&sy[e*800])[q];
    float ps = sv.x + sv.y + sv.z + sv.w;
    ps += __shfl_xor(ps, 1); ps += __shfl_xor(ps, 2);
    float mu = ps * (1.f/16.f);
    v4f dv = {sv.x-mu, sv.y-mu, sv.z-mu, sv.w-mu};
    float pv = dv.x*dv.x + dv.y*dv.y + dv.z*dv.z + dv.w*dv.w;
    pv += __shfl_xor(pv, 1); pv += __shfl_xor(pv, 2);
    float rinv = 1.f/sqrtf(pv*(1.f/16.f) + 1e-5f);

    v4f g0 = ((const v4f*)gamma)[q];   // gamma[0][0][4q..]
    v4f bt = ((const v4f*)beta)[q];
    v4f fm4, fp4;
    #pragma unroll
    for (int i = 0; i < 4; i++){
      float sh = dv[i]*rinv*g0[i] + bt[i];
      float sp = (sh > 20.f) ? sh : __logf(1.f + __expf(sh));
      float th = tanhf(sp);
      float sg = 1.f/(1.f + __expf(-sh));
      fm4[i] = sh*th;
      fp4[i] = th + sh*(1.f - th*th)*sg;
    }

    // rewrite: blk * (gamma[p][l]/rms) * fp
    #pragma unroll
    for (int l = 0; l < 5; l++){
      v4f g = ((const v4f*)(gamma + p*80 + l*16))[q];
      v4f sc = {invr[l]*g.x*fp4.x, invr[l]*g.y*fp4.y, invr[l]*g.z*fp4.z, invr[l]*g.w*fp4.w};
      for (int m = l*l; m < (l+1)*(l+1); m++){
        v4f* rp = &((v4f*)(base + m*16))[q];
        v4f v = *rp;
        *rp = v4f{v.x*sc.x, v.y*sc.y, v.z*sc.z, v.w*sc.w};
      }
    }
    if (p == 0) ((v4f*)&sy[e*800])[q] = fm4;   // scalar -> mish value
  }
  __syncthreads();

  // ---- edense #2 + residual (WMMA, residual via C operand) ----
  edense_y<true>(sy, W2, b2, c, h, w >> 1, w & 1);
  __syncthreads();

  // ---- basis + edense_b (WMMA): 9 tiles round-robin over waves ----
  {
    int bond = e0 + c; int bi = (bond < nBonds) ? bond : (nBonds > 0 ? nBonds - 1 : 0);
    float dx = disp[3*bi], dy = disp[3*bi+1], dz = disp[3*bi+2];
    float r  = sqrtf(dx*dx + dy*dy + dz*dz);
    float ir = 1.f/(r + 1e-12f);
    float x = dx*ir, y = dy*ir, z = dz*ir;
    float Y[9];
    Y[0] = 0.28209479177387814f;
    Y[1] = 0.4886025119029199f*y;
    Y[2] = 0.4886025119029199f*z;
    Y[3] = 0.4886025119029199f*x;
    Y[4] = 1.0925484305920792f*x*y;
    Y[5] = 1.0925484305920792f*y*z;
    Y[6] = 0.31539156525252005f*(3.f*z*z - 1.f);
    Y[7] = 1.0925484305920792f*x*z;
    Y[8] = 0.5462742152960396f*(x*x - y*y);
    float cutv = (r < 5.f) ? 0.5f*(__cosf(0.6283185307179586f*r) + 1.f) : 0.f;
    _Float16 rbh[8];
    #pragma unroll
    for (int i = 0; i < 8; i++){
      float ctr = (float)(8*h + i)*(5.f/15.f);   // linspace(0,5,16)
      float d = (r - ctr)*3.2f;                  // 1/width = F/CUTOFF
      rbh[i] = (_Float16)(__expf(-d*d)*cutv);
    }
    for (int n = w; n < 9; n += 4){
      int l = (n == 0) ? 0 : (n < 4 ? 1 : 2);
      v16h B;
      #pragma unroll
      for (int i = 0; i < 16; i++){
        float wv = (i < 8) ? Wb[(l*16 + 8*h + i)*16 + c] : 0.f;
        B[i] = (_Float16)wv;
      }
      v16h A;
      _Float16 yh = (_Float16)Y[n];
      #pragma unroll
      for (int i = 0; i < 8; i++){ A[i] = yh*rbh[i]; A[8+i] = (_Float16)0.f; }
      v8f C;
      float bv = (n == 0) ? bb[c] : 0.f;
      #pragma unroll
      for (int r2 = 0; r2 < 8; r2++) C[r2] = bv;
      v8f D = __builtin_amdgcn_wmma_f32_16x16x32_f16(false, A, false, B,
                                                     (short)0, C, false, false);
      #pragma unroll
      for (int r2 = 0; r2 < 8; r2++)
        sbed[((r2 + 8*h)*9 + n)*16 + c] = D[r2];
    }
  }
  __syncthreads();

  // ---- tensor product: thread handles (bond=t&15, f = 2*(t>>4) + fi) ----
  // Result is written back IN PLACE into the same (bond,f) column of sy that
  // this item (and only this item) reads -> no cross-item hazard.
  {
    const int e = t & 15;
    const int fg = t >> 4;         // 0..7
    #pragma unroll 1
    for (int fi = 0; fi < 2; fi++){
      const int f = 2*fg + fi;
      float a[9];
      #pragma unroll
      for (int n = 0; n < 9; n++) a[n] = sbed[(e*9 + n)*16 + f];
      float b[2][25];
      #pragma unroll
      for (int p = 0; p < 2; p++)
        #pragma unroll
        for (int m = 0; m < 25; m++) b[p][m] = sy[((e*2 + p)*25 + m)*16 + f];
      float o[2][25];
      #pragma unroll
      for (int p = 0; p < 2; p++)
        #pragma unroll
        for (int k = 0; k < 25; k++) o[p][k] = 0.f;

      // PATHS in exact reference order
      tp_path< 0,0,0,0>(Wtp,f,a,b,o); tp_path< 1,0,1,1>(Wtp,f,a,b,o);
      tp_path< 2,0,2,2>(Wtp,f,a,b,o); tp_path< 3,0,3,3>(Wtp,f,a,b,o);
      tp_path< 4,0,4,4>(Wtp,f,a,b,o);
      tp_path< 5,1,0,1>(Wtp,f,a,b,o); tp_path< 6,1,1,0>(Wtp,f,a,b,o);
      tp_path< 7,1,1,1>(Wtp,f,a,b,o); tp_path< 8,1,1,2>(Wtp,f,a,b,o);
      tp_path< 9,1,2,1>(Wtp,f,a,b,o); tp_path<10,1,2,2>(Wtp,f,a,b,o);
      tp_path<11,1,2,3>(Wtp,f,a,b,o); tp_path<12,1,3,2>(Wtp,f,a,b,o);
      tp_path<13,1,3,3>(Wtp,f,a,b,o); tp_path<14,1,3,4>(Wtp,f,a,b,o);
      tp_path<15,1,4,3>(Wtp,f,a,b,o); tp_path<16,1,4,4>(Wtp,f,a,b,o);
      tp_path<17,2,0,2>(Wtp,f,a,b,o); tp_path<18,2,1,1>(Wtp,f,a,b,o);
      tp_path<19,2,1,2>(Wtp,f,a,b,o); tp_path<20,2,1,3>(Wtp,f,a,b,o);
      tp_path<21,2,2,0>(Wtp,f,a,b,o); tp_path<22,2,2,1>(Wtp,f,a,b,o);
      tp_path<23,2,2,2>(Wtp,f,a,b,o); tp_path<24,2,2,3>(Wtp,f,a,b,o);
      tp_path<25,2,2,4>(Wtp,f,a,b,o); tp_path<26,2,3,1>(Wtp,f,a,b,o);
      tp_path<27,2,3,2>(Wtp,f,a,b,o); tp_path<28,2,3,3>(Wtp,f,a,b,o);
      tp_path<29,2,3,4>(Wtp,f,a,b,o); tp_path<30,2,4,2>(Wtp,f,a,b,o);
      tp_path<31,2,4,3>(Wtp,f,a,b,o); tp_path<32,2,4,4>(Wtp,f,a,b,o);

      #pragma unroll
      for (int p = 0; p < 2; p++)
        #pragma unroll
        for (int k = 0; k < 25; k++)
          sy[((e*2 + p)*25 + k)*16 + f] = o[p][k];
    }
  }
  __syncthreads();   // all tp DS stores committed before the LDS->global flush

  // ---- coalesced output flush: LDS -> HBM ----
  // 16 bonds x 800 f32 contiguous; async DMA from LDS when available.
  {
    v4f* out4 = (v4f*)outp;
    v4f* sy4  = (v4f*)sy;
    for (int tt = t; tt < 3200; tt += 128){
      int e = tt / 200, q = tt - e*200;
      int bond = e0 + e;
      if (bond < nBonds){
#ifdef HAVE_ASYNC_LDS
        __builtin_amdgcn_global_store_async_from_lds_b128(
            (g_v4i*)(out4 + (size_t)bond*200 + q),
            (l_v4i*)(sy4 + tt), 0, 0);
#else
        out4[(size_t)bond*200 + q] = sy4[tt];
#endif
      }
    }
#ifdef HAVE_ASYNC_LDS
#if __has_builtin(__builtin_amdgcn_s_wait_asynccnt)
    __builtin_amdgcn_s_wait_asynccnt(0);
#else
    asm volatile("s_wait_asynccnt 0" ::: "memory");
#endif
#endif
  }
}

// ====================== launch ======================
extern "C" void kernel_launch(void* const* d_in, const int* in_sizes, int n_in,
                              void* d_out, int out_size, void* d_ws, size_t ws_size,
                              hipStream_t stream) {
  (void)n_in; (void)out_size; (void)d_ws; (void)ws_size;
  const float* desc  = (const float*)d_in[0];
  const int*   nidx  = (const int*)  d_in[1];
  const float* disp  = (const float*)d_in[2];
  const float* W1    = (const float*)d_in[3];
  const float* b1    = (const float*)d_in[4];
  const float* gamma = (const float*)d_in[5];
  const float* beta  = (const float*)d_in[6];
  const float* W2    = (const float*)d_in[7];
  const float* b2    = (const float*)d_in[8];
  const float* Wb    = (const float*)d_in[9];
  const float* bb    = (const float*)d_in[10];
  const float* Wtp   = (const float*)d_in[11];

  int nBonds  = in_sizes[1] / 2;
  int nBlocks = (nBonds + 15) / 16;
  bctmd_kernel<<<dim3(nBlocks), dim3(128), 0, stream>>>(
      desc, nidx, disp, W1, b1, gamma, beta, W2, b2, Wb, bb, Wtp,
      (float*)d_out, nBonds);
}
